// GroupedQueryAttention_22668837388551
// MI455X (gfx1250) — matrix operand order repaired and checked by reading, not compiled, not run on hardware
//
#include <hip/hip_runtime.h>
#include <hip/hip_bf16.h>

typedef __attribute__((ext_vector_type(16))) _Float16 v16h;
typedef __attribute__((ext_vector_type(8)))  _Float16 v8h;
typedef __attribute__((ext_vector_type(8)))  float    v8f;

union AFrag { v16h v; v8h h[2]; };

#define SEQ    2048
#define DMODEL 1024
#define DKVTOT 256   // N_GROUPS * D_K
#define DK     64
#define NHEADS 16

// ---- 16-lane-half reductions (rows of a 16x16 C/D tile live in one half) ----
__device__ __forceinline__ float half_max16(float v) {
  v = fmaxf(v, __shfl_xor(v, 1));
  v = fmaxf(v, __shfl_xor(v, 2));
  v = fmaxf(v, __shfl_xor(v, 4));
  v = fmaxf(v, __shfl_xor(v, 8));
  return v;
}
__device__ __forceinline__ float half_sum16(float v) {
  v += __shfl_xor(v, 1);
  v += __shfl_xor(v, 2);
  v += __shfl_xor(v, 4);
  v += __shfl_xor(v, 8);
  return v;
}

// Packed A-fragment offset for matrix element (row, col), K = row length.
// Fragment block = 512 halves: [mtile][kchunk][lane][16]; lane = 16*lh + row%16,
// half j: col%32 = 16*(j/8) + 8*lh + (j%8).
__device__ __forceinline__ size_t apack_off(int row, int col, int K) {
  int mt = row >> 4, ll = row & 15;
  int kc = col >> 5, c32 = col & 31;
  int lane = (((c32 >> 3) & 1) << 4) + ll;
  int j = (c32 & 7) | ((c32 >> 4) << 3);
  return (((size_t)mt * (K >> 5) + kc) * 32 + lane) * 16 + j;
}

// ---- Kernel 1: RoPE on X -> f16 directly in packed A-fragment layout ----
__global__ void rope_pack_kernel(const float* __restrict__ X, _Float16* __restrict__ XrP) {
  const int half = DMODEL / 2;
  int idx = blockIdx.x * blockDim.x + threadIdx.x;
  if (idx >= SEQ * half) return;
  int s = idx / half;
  int i = idx - s * half;
  float theta = expf(-(float)i * (logf(10000.0f) / (float)half));
  float ang = (float)s * theta;
  float cs = cosf(ang), sn = sinf(ang);
  float xe = X[s * DMODEL + 2 * i];
  float xo = X[s * DMODEL + 2 * i + 1];
  XrP[apack_off(s, i,        DMODEL)] = (_Float16)(xe * cs - xo * sn);
  XrP[apack_off(s, half + i, DMODEL)] = (_Float16)(xe * sn + xo * cs);
}

// ---- Kernel 2: repack row-major f16 [M x K] into packed A-fragment layout ----
__global__ void apack_kernel(const _Float16* __restrict__ A, _Float16* __restrict__ Ap,
                             int M, int K) {
  int idx = blockIdx.x * blockDim.x + threadIdx.x;
  if (idx >= M * K) return;
  int j = idx & 15, lane = (idx >> 4) & 31;
  int e = idx >> 9;
  int kc = e % (K >> 5), mt = e / (K >> 5);
  int row = (mt << 4) + (lane & 15);
  int col = (kc << 5) + (((lane >> 4) & 1) << 3) + (j & 7) + ((j >> 3) << 4);
  Ap[idx] = A[(size_t)row * K + col];
}

// ---- Kernel 3: W[N x K] f32 -> packed B-fragment f16. Block = [kchunk][ntile][lane][16];
// lane L: N = ntile*16 + L%16, half j: K = kchunk*32 + 8*(L/16) + (j%8) + 16*(j/8).
__global__ void wpack_kernel(const float* __restrict__ W, _Float16* __restrict__ Bp,
                             int N, int K) {
  int idx = blockIdx.x * blockDim.x + threadIdx.x;
  if (idx >= N * K) return;
  int j = idx & 15, lane = (idx >> 4) & 31;
  int e = idx >> 9;
  int nt = e % (N >> 4), kc = e / (N >> 4);
  int Kidx = (kc << 5) + ((lane >> 4) << 3) + (j & 7) + ((j >> 3) << 4);
  int Nidx = (nt << 4) + (lane & 15);
  Bp[idx] = (_Float16)W[(size_t)Nidx * K + Kidx];
}

// ---- Kernel 4: Kh [S x 256] -> packed B-fragments for QK^T.
// Block = [g][kt16][chunk][lane][16]; lane L: key = kt16*16 + L%16,
// half j: dim = g*64 + chunk*32 + 8*(L/16) + (j%8) + 16*(j/8).
__global__ void kpack_kernel(const _Float16* __restrict__ Kh, _Float16* __restrict__ Kp) {
  int idx = blockIdx.x * blockDim.x + threadIdx.x;
  if (idx >= SEQ * DKVTOT) return;
  int j = idx & 15, lane = (idx >> 4) & 31;
  int e = idx >> 9;
  int c = e & 1; e >>= 1;
  int kt = e % (SEQ >> 4), g = e / (SEQ >> 4);
  int dim = g * DK + (c << 5) + ((lane >> 4) << 3) + (j & 7) + ((j >> 3) << 4);
  int key = (kt << 4) + (lane & 15);
  Kp[idx] = Kh[(size_t)key * DKVTOT + dim];
}

// ---- Kernel 5: Vh [S x 256] -> packed B-fragments for P*V.
// Block = [g][kt32][dt][lane][16]; lane L: dim = g*64 + dt*16 + L%16,
// half j: key = kt32*32 + 8*(L/16) + (j%8) + 16*(j/8).
__global__ void vpack_kernel(const _Float16* __restrict__ Vh, _Float16* __restrict__ Vp) {
  int idx = blockIdx.x * blockDim.x + threadIdx.x;
  if (idx >= SEQ * DKVTOT) return;
  int j = idx & 15, lane = (idx >> 4) & 31;
  int e = idx >> 9;
  int dt = e & 3; e >>= 2;
  int kt = e % (SEQ >> 5), g = e / (SEQ >> 5);
  int key = (kt << 5) + ((lane >> 4) << 3) + (j & 7) + ((j >> 3) << 4);
  int dim = g * DK + (dt << 4) + (lane & 15);
  Vp[idx] = Vh[(size_t)key * DKVTOT + dim];
}

// ---- Kernel 6: WMMA GEMM, C[M x N] = A*B, packed fragments, 16x64 tile per wave ----
template <typename OutT>
__global__ void __launch_bounds__(32)
gemm_wmma_kernel(const _Float16* __restrict__ Ap, const _Float16* __restrict__ Bp,
                 OutT* __restrict__ C, int N, int K) {
  int lane = threadIdx.x;
  int lh = lane >> 4, ll = lane & 15;
  int mt = blockIdx.x;            // 16-row tile
  int nt4 = blockIdx.y;           // 64-col tile
  int nTiles = N >> 4, kChunks = K >> 5;
  v8f acc[4];
  #pragma unroll
  for (int t = 0; t < 4; ++t) acc[t] = (v8f){};
  const _Float16* abase = Ap + ((size_t)mt * kChunks) * 512 + lane * 16;
  const _Float16* bbase = Bp + ((size_t)nt4 * 4) * 512 + lane * 16;
  for (int kc = 0; kc < kChunks; ++kc) {
    v16h a = *(const v16h*)(abase + (size_t)kc * 512);
    #pragma unroll
    for (int t = 0; t < 4; ++t) {
      v16h b = *(const v16h*)(bbase + ((size_t)kc * nTiles + t) * 512);
      acc[t] = __builtin_amdgcn_wmma_f32_16x16x32_f16(false, a, false, b, (short)0, acc[t],
                                                      false, false);
    }
  }
  #pragma unroll
  for (int t = 0; t < 4; ++t)
    #pragma unroll
    for (int r = 0; r < 8; ++r) {
      int row = (mt << 4) + r + 8 * lh;        // C/D layout
      C[(size_t)row * N + (nt4 << 6) + (t << 4) + ll] = (OutT)acc[t][r];
    }
}

// ---- Kernel 7: flash attention, one wave = (head, 16 queries), 32-key tiles ----
__global__ void __launch_bounds__(32)
attn_wmma_kernel(const _Float16* __restrict__ Qh, const _Float16* __restrict__ Kp,
                 const _Float16* __restrict__ Vp, _Float16* __restrict__ Oh) {
  __shared__ __align__(32) _Float16 sP[16 * 32];
  int lane = threadIdx.x;
  int lh = lane >> 4, ll = lane & 15;
  int qBase = blockIdx.x * 16;
  int h = blockIdx.y;
  int g = h >> 2;

  // Q as two A-fragments (d_k = 64 = 2 chunks of 32), from row-major Qh
  const _Float16* qrow = Qh + (size_t)(qBase + ll) * DMODEL + h * DK + 8 * lh;
  AFrag qa0, qa1;
  qa0.h[0] = *(const v8h*)(qrow);      qa0.h[1] = *(const v8h*)(qrow + 16);
  qa1.h[0] = *(const v8h*)(qrow + 32); qa1.h[1] = *(const v8h*)(qrow + 48);

  v8f o[4];
  float mrow[8], lrow[8];
  #pragma unroll
  for (int dt = 0; dt < 4; ++dt) o[dt] = (v8f){};
  #pragma unroll
  for (int r = 0; r < 8; ++r) { mrow[r] = -1e30f; lrow[r] = 0.0f; }

  const _Float16* kgrp = Kp + (size_t)g * (SEQ >> 4) * 2 * 512 + lane * 16;
  const _Float16* vgrp = Vp + (size_t)g * (SEQ >> 5) * 4 * 512 + lane * 16;

  for (int kBase = 0; kBase <= qBase + 15; kBase += 32) {
    // ---- S = Q*K^T for 32 keys (two 16-key tiles); packed, coalesced fragments ----
    float sv[2][8];
    #pragma unroll
    for (int nt = 0; nt < 2; ++nt) {
      int kt16 = (kBase >> 4) + nt;
      v8f s = {};
      v16h b0 = *(const v16h*)(kgrp + ((size_t)kt16 * 2 + 0) * 512);
      v16h b1 = *(const v16h*)(kgrp + ((size_t)kt16 * 2 + 1) * 512);
      s = __builtin_amdgcn_wmma_f32_16x16x32_f16(false, qa0.v, false, b0, (short)0, s,
                                                 false, false);
      s = __builtin_amdgcn_wmma_f32_16x16x32_f16(false, qa1.v, false, b1, (short)0, s,
                                                 false, false);
      #pragma unroll
      for (int r = 0; r < 8; ++r) {
        int q = qBase + r + 8 * lh;
        int k = kBase + nt * 16 + ll;
        float val = s[r] * 0.125f;                   // 1/sqrt(64)
        sv[nt][r] = (k <= q) ? val : -1e30f;         // causal mask
      }
    }
    // ---- online softmax ----
    float alpha[8], pr0[8], pr1[8];
    #pragma unroll
    for (int r = 0; r < 8; ++r) {
      float lm = half_max16(fmaxf(sv[0][r], sv[1][r]));
      float mnew = fmaxf(mrow[r], lm);
      alpha[r] = __expf(mrow[r] - mnew);
      float p0 = __expf(sv[0][r] - mnew);
      float p1 = __expf(sv[1][r] - mnew);
      lrow[r] = lrow[r] * alpha[r] + half_sum16(p0 + p1);
      mrow[r] = mnew;
      pr0[r] = p0; pr1[r] = p1;
    }
    #pragma unroll
    for (int dt = 0; dt < 4; ++dt)
      #pragma unroll
      for (int r = 0; r < 8; ++r) o[dt][r] *= alpha[r];

    // ---- P: C/D layout -> A-fragment layout via LDS (DS in-order within a wave) ----
    #pragma unroll
    for (int r = 0; r < 8; ++r) {
      int row = r + 8 * lh;
      sP[row * 32 + ll]      = (_Float16)pr0[r];
      sP[row * 32 + 16 + ll] = (_Float16)pr1[r];
    }
    AFrag pa;
    const _Float16* pp = &sP[ll * 32 + 8 * lh];
    pa.h[0] = *(const v8h*)(pp);
    pa.h[1] = *(const v8h*)(pp + 16);

    // ---- O += P*V (4 dim tiles), packed coalesced V fragments ----
    int kt32 = kBase >> 5;
    #pragma unroll
    for (int dt = 0; dt < 4; ++dt) {
      v16h vb = *(const v16h*)(vgrp + ((size_t)kt32 * 4 + dt) * 512);
      o[dt] = __builtin_amdgcn_wmma_f32_16x16x32_f16(false, pa.v, false, vb, (short)0, o[dt],
                                                     false, false);
    }
  }

  // ---- normalize and store row-major (repacked before the W_o GEMM) ----
  #pragma unroll
  for (int r = 0; r < 8; ++r) {
    float inv = 1.0f / lrow[r];
    int row = qBase + r + 8 * lh;
    #pragma unroll
    for (int dt = 0; dt < 4; ++dt)
      Oh[(size_t)row * DMODEL + h * DK + dt * 16 + ll] = (_Float16)(o[dt][r] * inv);
  }
}

extern "C" void kernel_launch(void* const* d_in, const int* in_sizes, int n_in,
                              void* d_out, int out_size, void* d_ws, size_t ws_size,
                              hipStream_t stream) {
  const float* X  = (const float*)d_in[0];
  const float* Wq = (const float*)d_in[1];
  const float* Wk = (const float*)d_in[2];
  const float* Wv = (const float*)d_in[3];
  const float* Wo = (const float*)d_in[4];
  float* out = (float*)d_out;

  _Float16* ws = (_Float16*)d_ws;
  size_t off = 0;
  _Float16* XrP = ws + off; off += (size_t)SEQ * DMODEL;       // packed A
  _Float16* WqP = ws + off; off += (size_t)DMODEL * DMODEL;    // packed B
  _Float16* WkP = ws + off; off += (size_t)DMODEL * DKVTOT;
  _Float16* WvP = ws + off; off += (size_t)DMODEL * DKVTOT;
  _Float16* WoP = ws + off; off += (size_t)DMODEL * DMODEL;
  _Float16* Qh  = ws + off; off += (size_t)SEQ * DMODEL;       // row-major
  _Float16* Kh  = ws + off; off += (size_t)SEQ * DKVTOT;
  _Float16* Vh  = ws + off; off += (size_t)SEQ * DKVTOT;
  _Float16* Kp  = ws + off; off += (size_t)SEQ * DKVTOT;       // packed B frags
  _Float16* Vp  = ws + off; off += (size_t)SEQ * DKVTOT;
  _Float16* Ah  = ws + off; off += (size_t)SEQ * DMODEL;       // row-major
  _Float16* AhP = ws + off; off += (size_t)SEQ * DMODEL;       // packed A  (~26 MB total)

  // 1) RoPE + convert straight into packed A layout
  rope_pack_kernel<<<(SEQ * (DMODEL / 2) + 255) / 256, 256, 0, stream>>>(X, XrP);
  // 2) weights -> packed B fragments
  wpack_kernel<<<(DMODEL * DMODEL + 255) / 256, 256, 0, stream>>>(Wq, WqP, DMODEL, DMODEL);
  wpack_kernel<<<(DKVTOT * DMODEL + 255) / 256, 256, 0, stream>>>(Wk, WkP, DKVTOT, DMODEL);
  wpack_kernel<<<(DKVTOT * DMODEL + 255) / 256, 256, 0, stream>>>(Wv, WvP, DKVTOT, DMODEL);
  wpack_kernel<<<(DMODEL * DMODEL + 255) / 256, 256, 0, stream>>>(Wo, WoP, DMODEL, DMODEL);
  // 3) projections (WMMA, 16x64 tile per wave)
  gemm_wmma_kernel<_Float16><<<dim3(SEQ / 16, DMODEL / 64), 32, 0, stream>>>(XrP, WqP, Qh, DMODEL, DMODEL);
  gemm_wmma_kernel<_Float16><<<dim3(SEQ / 16, DKVTOT / 64), 32, 0, stream>>>(XrP, WkP, Kh, DKVTOT, DMODEL);
  gemm_wmma_kernel<_Float16><<<dim3(SEQ / 16, DKVTOT / 64), 32, 0, stream>>>(XrP, WvP, Vh, DKVTOT, DMODEL);
  // 4) K/V -> attention fragment layouts
  kpack_kernel<<<(SEQ * DKVTOT + 255) / 256, 256, 0, stream>>>(Kh, Kp);
  vpack_kernel<<<(SEQ * DKVTOT + 255) / 256, 256, 0, stream>>>(Vh, Vp);
  // 5) flash attention (all-WMMA inner loop)
  attn_wmma_kernel<<<dim3(SEQ / 16, NHEADS), 32, 0, stream>>>(Qh, Kp, Vp, Ah);
  // 6) output projection
  apack_kernel<<<(SEQ * DMODEL + 255) / 256, 256, 0, stream>>>(Ah, AhP, SEQ, DMODEL);
  gemm_wmma_kernel<float><<<dim3(SEQ / 16, DMODEL / 64), 32, 0, stream>>>(AhP, WoP, out, DMODEL, DMODEL);
}
